// SDPA_63891933495303
// MI455X (gfx1250) — matrix operand order repaired and checked by reading, not compiled, not run on hardware
//
#include <hip/hip_runtime.h>

typedef __attribute__((ext_vector_type(16))) _Float16 v16h;
typedef __attribute__((ext_vector_type(8)))  _Float16 v8h;
typedef __attribute__((ext_vector_type(2)))  _Float16 h2;
typedef __attribute__((ext_vector_type(8)))  float    v8f;

#define Bb   2
#define Hh   16
#define Ss   2048
#define DK   64
#define DV   64
#define BM   128     // q rows per block (8 waves x 16)
#define BN   64      // keys per iteration
#define NW   8       // waves per block
#define LDST 72      // padded LDS row stride (halves); 144B rows -> 16B aligned

#define LOG2E 1.4426950408889634f

// ---- single-instruction DPP16 reduction hops (rotate within 16-lane row) ----
// v_max_num_f32_dpp d, s0, s1 : DPP applies to s0 -> d = max(ror(x), x)
#define ROWMAX_HOP(x, N)                                                      \
  asm("v_max_num_f32_dpp %0, %0, %0 row_ror:" #N                              \
      " row_mask:0xf bank_mask:0xf bound_ctrl:1" : "+v"(x))
#define ROWSUM_HOP(x, N)                                                      \
  asm("v_add_f32_dpp %0, %0, %0 row_ror:" #N                                  \
      " row_mask:0xf bank_mask:0xf bound_ctrl:1" : "+v"(x))

__device__ __forceinline__ float rowmax16(float x) {
  ROWMAX_HOP(x, 8); ROWMAX_HOP(x, 4); ROWMAX_HOP(x, 2); ROWMAX_HOP(x, 1);
  return x;
}
__device__ __forceinline__ float rowsum16(float x) {
  ROWSUM_HOP(x, 8); ROWSUM_HOP(x, 4); ROWSUM_HOP(x, 2); ROWSUM_HOP(x, 1);
  return x;
}

__global__ __launch_bounds__(256, 1)
void sdpa_fa_f16wmma(const float* __restrict__ Q, const float* __restrict__ K,
                     const float* __restrict__ V, const float* __restrict__ M,
                     float* __restrict__ O) {
  __shared__ _Float16 Kl[BN * LDST];          // [key][dk]
  __shared__ _Float16 Vt[DV * LDST];          // [dv][key]  (transposed V)
  __shared__ _Float16 Pl[NW][16 * LDST];      // per-wave P staging

  const int tid  = threadIdx.x;
  const int lane = tid & 31;
  const int w    = tid >> 5;
  const int m16  = lane & 15;
  const int hi   = lane >> 4;          // 0: rows r, 1: rows r+8 (C-layout halves)

  const int bh    = blockIdx.y;
  const int qBlk  = blockIdx.x * BM;
  const int qRow0 = qBlk + w * 16;

  const float* Qb = Q + (size_t)bh * Ss * DK;
  const float* Kb = K + (size_t)bh * Ss * DK;
  const float* Vb = V + (size_t)bh * Ss * DV;
  float*       Ob = O + (size_t)bh * Ss * DV;

  // ---- Q A-fragments; scale = 1/sqrt(64) * log2(e) folded in (exp2 domain) ----
  v16h qa0, qa1;
  {
    const float sc = 0.125f * LOG2E;
    const float* qrow = Qb + (size_t)(qRow0 + m16) * DK;
    #pragma unroll
    for (int v = 0; v < 8; ++v) {
      int k = (v >> 2) * 16 + (v & 3) * 2 + hi * 8;
      qa0[2*v]   = (_Float16)(qrow[k]      * sc);
      qa0[2*v+1] = (_Float16)(qrow[k + 1]  * sc);
      qa1[2*v]   = (_Float16)(qrow[32 + k]     * sc);
      qa1[2*v+1] = (_Float16)(qrow[32 + k + 1] * sc);
    }
  }

  float mrow[8], lrow[8];
  v8f oacc[4];
  #pragma unroll
  for (int r = 0; r < 8; ++r) { mrow[r] = -1e30f; lrow[r] = 0.0f; }
  #pragma unroll
  for (int t = 0; t < 4; ++t) { v8f z = {}; oacc[t] = z; }

  // staging assignment: row-pair (2 rows) x 8 cols per thread
  const int rp = tid >> 3;        // 0..31 -> rows 2rp, 2rp+1
  const int c0 = (tid & 7) * 8;   // column base

  for (int j = 0; j < Ss / BN; ++j) {
    __syncthreads();
    // ---- cooperative stage: K -> Kl[key][dk] (b128), V -> Vt[dv][key] (b32 pairs) ----
    {
      const float* k0p = Kb + (size_t)(j * BN + 2 * rp) * DK + c0;
      const float* v0p = Vb + (size_t)(j * BN + 2 * rp) * DV + c0;
      #pragma unroll
      for (int rr = 0; rr < 2; ++rr) {
        const float4* kq = (const float4*)(k0p + rr * DK);
        float4 a = kq[0], b = kq[1];
        v8h kh;
        kh[0] = (_Float16)a.x; kh[1] = (_Float16)a.y;
        kh[2] = (_Float16)a.z; kh[3] = (_Float16)a.w;
        kh[4] = (_Float16)b.x; kh[5] = (_Float16)b.y;
        kh[6] = (_Float16)b.z; kh[7] = (_Float16)b.w;
        *(v8h*)&Kl[(2 * rp + rr) * LDST + c0] = kh;
      }
      {
        const float4* vq0 = (const float4*)(v0p);
        const float4* vq1 = (const float4*)(v0p + DV);
        float4 a0 = vq0[0], a1 = vq0[1];
        float4 b0 = vq1[0], b1 = vq1[1];
        float r0[8] = {a0.x, a0.y, a0.z, a0.w, a1.x, a1.y, a1.z, a1.w};
        float r1[8] = {b0.x, b0.y, b0.z, b0.w, b1.x, b1.y, b1.z, b1.w};
        #pragma unroll
        for (int c = 0; c < 8; ++c) {
          h2 p; p[0] = (_Float16)r0[c]; p[1] = (_Float16)r1[c];
          *(h2*)&Vt[(c0 + c) * LDST + 2 * rp] = p;
        }
      }
      if (j + 1 < Ss / BN) {   // global_prefetch_b8 for next tile
        __builtin_prefetch(Kb + (size_t)((j + 1) * BN + 2 * rp) * DK + c0, 0, 3);
        __builtin_prefetch(Vb + (size_t)((j + 1) * BN + 2 * rp) * DV + c0, 0, 3);
      }
    }
    __syncthreads();

    // ---- S = (Q*sc) @ K^T : 4 N-tiles x 2 K-chunks ----
    v8f s[4];
    #pragma unroll
    for (int t = 0; t < 4; ++t) {
      v8f acc = {};
      v16h b0, b1;
      const _Float16* kp = &Kl[(t * 16 + m16) * LDST];
      #pragma unroll
      for (int v = 0; v < 8; ++v) {
        int k0 = (v >> 2) * 16 + (v & 3) * 2 + hi * 8;
        b0[2*v] = kp[k0];      b0[2*v+1] = kp[k0 + 1];
        b1[2*v] = kp[32 + k0]; b1[2*v+1] = kp[32 + k0 + 1];
      }
      acc = __builtin_amdgcn_wmma_f32_16x16x32_f16(false, qa0, false, b0, (short)0, acc, false, false);
      acc = __builtin_amdgcn_wmma_f32_16x16x32_f16(false, qa1, false, b1, (short)0, acc, false, false);
      s[t] = acc;
    }

    // ---- additive mask, scaled into exp2 domain (fmaf -> v_fmamk) ----
    #pragma unroll
    for (int t = 0; t < 4; ++t) {
      int kcol = j * BN + t * 16 + m16;
      #pragma unroll
      for (int r = 0; r < 8; ++r) {
        int q = qRow0 + r + hi * 8;
        s[t][r] = fmaf(M[(size_t)q * Ss + kcol], LOG2E, s[t][r]);
      }
    }

    // ---- online softmax in base-2 domain; DPP16 single-op reduction hops ----
    float alpha[8];
    #pragma unroll
    for (int r = 0; r < 8; ++r) {
      float mx = fmaxf(fmaxf(s[0][r], s[1][r]), fmaxf(s[2][r], s[3][r]));
      mx = rowmax16(mx);
      float mnew = fmaxf(mrow[r], mx);
      alpha[r] = exp2f(mrow[r] - mnew);
      mrow[r]  = mnew;
      float rs = 0.0f;
      #pragma unroll
      for (int t = 0; t < 4; ++t) {
        float p = exp2f(s[t][r] - mnew);
        s[t][r] = p;
        rs += p;
      }
      rs = rowsum16(rs);
      lrow[r] = lrow[r] * alpha[r] + rs;
    }

    // ---- P: C-layout -> per-wave LDS tile -> A-layout (same-wave DS in-order) ----
    _Float16* pw = &Pl[w][0];
    #pragma unroll
    for (int t = 0; t < 4; ++t)
      #pragma unroll
      for (int r = 0; r < 8; ++r)
        pw[(r + hi * 8) * LDST + t * 16 + m16] = (_Float16)s[t][r];

    v16h pa0, pa1;
    {
      const _Float16* pr = &pw[m16 * LDST];
      #pragma unroll
      for (int v = 0; v < 8; ++v) {
        int k = (v >> 2) * 16 + (v & 3) * 2 + hi * 8;
        pa0[2*v] = pr[k];      pa0[2*v+1] = pr[k + 1];
        pa1[2*v] = pr[32 + k]; pa1[2*v+1] = pr[32 + k + 1];
      }
    }

    // ---- O = O*alpha + P @ V ----
    #pragma unroll
    for (int t = 0; t < 4; ++t) {
      #pragma unroll
      for (int r = 0; r < 8; ++r) oacc[t][r] *= alpha[r];
      v16h b0, b1;
      const _Float16* vp = &Vt[(t * 16 + m16) * LDST];
      #pragma unroll
      for (int v = 0; v < 8; ++v) {
        int k0 = (v >> 2) * 16 + (v & 3) * 2 + hi * 8;
        b0[2*v] = vp[k0];      b0[2*v+1] = vp[k0 + 1];
        b1[2*v] = vp[32 + k0]; b1[2*v+1] = vp[32 + k0 + 1];
      }
      oacc[t] = __builtin_amdgcn_wmma_f32_16x16x32_f16(false, pa0, false, b0, (short)0, oacc[t], false, false);
      oacc[t] = __builtin_amdgcn_wmma_f32_16x16x32_f16(false, pa1, false, b1, (short)0, oacc[t], false, false);
    }
  }

  // ---- epilogue: divide by l, store fp32 ----
  #pragma unroll
  for (int r = 0; r < 8; ++r) {
    float inv = 1.0f / lrow[r];
    int q = qRow0 + r + hi * 8;
    float* orow = Ob + (size_t)q * DV;
    #pragma unroll
    for (int t = 0; t < 4; ++t)
      orow[t * 16 + m16] = oacc[t][r] * inv;
  }
}

extern "C" void kernel_launch(void* const* d_in, const int* in_sizes, int n_in,
                              void* d_out, int out_size, void* d_ws, size_t ws_size,
                              hipStream_t stream) {
  (void)in_sizes; (void)n_in; (void)out_size; (void)d_ws; (void)ws_size;
  const float* Q = (const float*)d_in[0];
  const float* K = (const float*)d_in[1];
  const float* V = (const float*)d_in[2];
  const float* M = (const float*)d_in[3];
  float*       O = (float*)d_out;

  dim3 grid(Ss / BM, Bb * Hh);
  dim3 block(32 * NW);
  sdpa_fa_f16wmma<<<grid, block, 0, stream>>>(Q, K, V, M, O);
}
